// MoE_1992864825975
// MI455X (gfx1250) — compile-verified, matrix operand order checked by reading, not run on hardware
//
#include <hip/hip_runtime.h>
#include <hip/hip_bf16.h>

// ---------------------------------------------------------------------------
// MoE (top-2 of 8 experts + shared expert) for gfx1250 / MI455X.
// bf16 WMMA (v_wmma_f32_16x16x32_bf16) GEMM pipeline, wave32, 8 waves/block,
// double-buffered LDS tiles (one barrier per K-step).
// ---------------------------------------------------------------------------

constexpr int kBS = 4, kSLEN = 2048, kDIM = 2048, kHID = 1792, kE = 8, kTOPK = 2;
constexpr int kT   = kBS * kSLEN;        // 8192 tokens
constexpr int kRR  = kT * kTOPK;         // 16384 routed rows
constexpr int BM = 128, BN = 64, BK = 32;
constexpr int GN = 128;                  // gemm2 N-tile (32x64 per wave)
constexpr int kMAXPAD = kRR + kE * BM;   // 17408 (each expert padded to 128)
constexpr int LDT = BK + 8;              // LDS row stride in bf16 (80B, 16B aligned)

typedef __attribute__((ext_vector_type(16))) __bf16 v16bf;
typedef __attribute__((ext_vector_type(8)))  __bf16 v8bf;
typedef __attribute__((ext_vector_type(8)))  float  v8f;

// ---------------------------------------------------------------------------
// Router: scores = softmax(x @ gate_w); biased top-2; per-expert counts.
// ---------------------------------------------------------------------------
__global__ void router_kernel(const float* __restrict__ x,
                              const float* __restrict__ gate_w,
                              const float* __restrict__ bias,
                              int* __restrict__ sel_e,
                              float* __restrict__ sel_s,
                              int* __restrict__ counts) {
  int t = blockIdx.x * blockDim.x + threadIdx.x;
  if (t >= kT) return;
  const float* xr = x + (size_t)t * kDIM;

  float logit[kE];
#pragma unroll
  for (int e = 0; e < kE; ++e) logit[e] = 0.f;
  for (int d = 0; d < kDIM; d += 4) {
    float4 xv = *(const float4*)(xr + d);
    const float* g = gate_w + (size_t)d * kE;
#pragma unroll
    for (int e = 0; e < kE; ++e)
      logit[e] += xv.x * g[e] + xv.y * g[kE + e] + xv.z * g[2 * kE + e] + xv.w * g[3 * kE + e];
  }
  float mx = logit[0];
#pragma unroll
  for (int e = 1; e < kE; ++e) mx = fmaxf(mx, logit[e]);
  float sc[kE];
  float den = 0.f;
#pragma unroll
  for (int e = 0; e < kE; ++e) { sc[e] = __expf(logit[e] - mx); den += sc[e]; }
  float inv = 1.f / den;
#pragma unroll
  for (int e = 0; e < kE; ++e) sc[e] *= inv;

  int i0 = 0; float b0 = -1e30f;
#pragma unroll
  for (int e = 0; e < kE; ++e) {
    float v = sc[e] + bias[e];
    if (v > b0) { b0 = v; i0 = e; }
  }
  int i1 = 0; float b1 = -1e30f;
#pragma unroll
  for (int e = 0; e < kE; ++e) {
    float v = sc[e] + bias[e];
    if (e != i0 && v > b1) { b1 = v; i1 = e; }
  }
  sel_e[t * 2 + 0] = i0; sel_s[t * 2 + 0] = sc[i0]; atomicAdd(&counts[i0], 1);
  sel_e[t * 2 + 1] = i1; sel_s[t * 2 + 1] = sc[i1]; atomicAdd(&counts[i1], 1);
}

__global__ void init_kernel(int* __restrict__ row_tok, int* __restrict__ counts) {
  int i = blockIdx.x * blockDim.x + threadIdx.x;
  if (i < kMAXPAD) row_tok[i] = -1;
  if (i < kE) counts[i] = 0;
}

__global__ void scan_kernel(const int* __restrict__ counts,
                            int* __restrict__ offs,
                            int* __restrict__ cursor) {
  if (threadIdx.x == 0 && blockIdx.x == 0) {
    int acc = 0;
    for (int e = 0; e < kE; ++e) {
      offs[e] = acc;
      cursor[e] = 0;
      acc += (counts[e] + BM - 1) / BM * BM;
    }
    offs[kE] = acc;
  }
}

__global__ void assign_kernel(const int* __restrict__ sel_e,
                              const float* __restrict__ sel_s,
                              const int* __restrict__ offs,
                              int* __restrict__ cursor,
                              int* __restrict__ row_tok,
                              float* __restrict__ row_scl) {
  int i = blockIdx.x * blockDim.x + threadIdx.x;
  if (i >= kRR) return;
  int e = sel_e[i];
  int slot = offs[e] + atomicAdd(&cursor[e], 1);
  row_tok[slot] = i / kTOPK;
  row_scl[slot] = sel_s[i];
}

// ---------------------------------------------------------------------------
// WMMA fragment fetch from an LDS tile (16 rows x 32 K, bf16, row stride LDT).
// 16-bit A-operand layout: lanes 0-15 hold K 0-7 / 16-23, lanes 16-31 hold
// K 8-15 / 24-31 (two 16B ds_load_b128).
// ---------------------------------------------------------------------------
__device__ __forceinline__ v16bf load_frag(const __bf16* lds) {
  int lane = threadIdx.x & 31;
  int m    = lane & 15;
  int klo  = (lane & 16) ? 8 : 0;
  const __bf16* p = lds + m * LDT;
  v8bf lo = *(const v8bf*)(p + klo);
  v8bf hi = *(const v8bf*)(p + klo + 16);
  v16bf r;
#pragma unroll
  for (int i = 0; i < 8; ++i) { r[i] = lo[i]; r[i + 8] = hi[i]; }
  return r;
}

// ---------------------------------------------------------------------------
// FFN first stage: H = silu(A @ W1) * (A @ W3), A rows gathered & scaled.
// offs == nullptr -> shared expert (identity rows, single weight).
// Block tile 128x64, wave tile 32x32 (x2 weight streams), double-buffered.
// ---------------------------------------------------------------------------
__global__ __launch_bounds__(256)
void ffn13_kernel(const float* __restrict__ X,
                  const float* __restrict__ W1base,
                  const float* __restrict__ W3base,
                  const int* __restrict__ row_tok,
                  const float* __restrict__ row_scl,
                  const int* __restrict__ offs,
                  __bf16* __restrict__ H) {
  __shared__ __align__(16) __bf16 As [2][BM][LDT];
  __shared__ __align__(16) __bf16 Bs1[2][BN][LDT];
  __shared__ __align__(16) __bf16 Bs3[2][BN][LDT];

  const int m0 = blockIdx.x * BM;
  const int n0 = blockIdx.y * BN;

  const float* W1 = W1base;
  const float* W3 = W3base;
  if (offs) {
    if (m0 >= offs[kE]) return;
    int e = 0;
    while (e + 1 < kE && m0 >= offs[e + 1]) ++e;
    W1 += (size_t)e * kDIM * kHID;
    W3 += (size_t)e * kDIM * kHID;
  }

  const int tid  = threadIdx.x;
  const int lane = tid & 31;
  const int wid  = tid >> 5;
  const int wm   = wid >> 1;   // 0..3 -> 32-row slice
  const int wn   = wid & 1;    // 0..1 -> 32-col slice

  v8f acc1[2][2], acc3[2][2];
#pragma unroll
  for (int i = 0; i < 2; ++i)
#pragma unroll
    for (int j = 0; j < 2; ++j) { v8f z = {}; acc1[i][j] = z; acc3[i][j] = z; }

  // Gather state: 2 threads per A row (16 fp32 each); 2 float4 per B stream.
  const int ar = tid >> 1;
  const int ac = (tid & 1) * 16;
  int tok = m0 + ar;
  float scale = 1.f;
  if (row_tok) {
    tok = row_tok[m0 + ar];
    if (tok >= 0) scale = row_scl[m0 + ar];
  }

  float4 aReg[4];
  float4 b1Reg[2], b3Reg[2];

  auto load_global = [&](int k0) {
    if (tok >= 0) {
      const float4* src = (const float4*)(X + (size_t)tok * kDIM + k0 + ac);
#pragma unroll
      for (int j = 0; j < 4; ++j) aReg[j] = src[j];
    } else {
      float4 z = {0.f, 0.f, 0.f, 0.f};
#pragma unroll
      for (int j = 0; j < 4; ++j) aReg[j] = z;
    }
#pragma unroll
    for (int f = 0; f < 2; ++f) {
      int fi = tid * 2 + f;            // 0..511
      int bk = fi >> 4;                // 0..31
      int bn = (fi & 15) * 4;          // 0..60
      const float* s1 = W1 + (size_t)(k0 + bk) * kHID + n0 + bn;
      const float* s3 = W3 + (size_t)(k0 + bk) * kHID + n0 + bn;
      b1Reg[f] = *(const float4*)s1;
      b3Reg[f] = *(const float4*)s3;
      __builtin_prefetch(s1 + (size_t)BK * kHID, 0, 1);  // k0 + 2*BK stream hint
      __builtin_prefetch(s3 + (size_t)BK * kHID, 0, 1);
    }
  };
  auto store_lds = [&](int p) {
#pragma unroll
    for (int j = 0; j < 4; ++j) {
      As[p][ar][ac + j * 4 + 0] = (__bf16)(aReg[j].x * scale);
      As[p][ar][ac + j * 4 + 1] = (__bf16)(aReg[j].y * scale);
      As[p][ar][ac + j * 4 + 2] = (__bf16)(aReg[j].z * scale);
      As[p][ar][ac + j * 4 + 3] = (__bf16)(aReg[j].w * scale);
    }
#pragma unroll
    for (int f = 0; f < 2; ++f) {
      int fi = tid * 2 + f;
      int bk = fi >> 4;
      int bn = (fi & 15) * 4;
      Bs1[p][bn + 0][bk] = (__bf16)b1Reg[f].x; Bs1[p][bn + 1][bk] = (__bf16)b1Reg[f].y;
      Bs1[p][bn + 2][bk] = (__bf16)b1Reg[f].z; Bs1[p][bn + 3][bk] = (__bf16)b1Reg[f].w;
      Bs3[p][bn + 0][bk] = (__bf16)b3Reg[f].x; Bs3[p][bn + 1][bk] = (__bf16)b3Reg[f].y;
      Bs3[p][bn + 2][bk] = (__bf16)b3Reg[f].z; Bs3[p][bn + 3][bk] = (__bf16)b3Reg[f].w;
    }
  };

  // Prologue: fill buffer 0.
  load_global(0);
  store_lds(0);
  __syncthreads();

  int p = 0;
  for (int k0 = 0; k0 < kDIM; k0 += BK) {
    const bool more = (k0 + BK) < kDIM;
    if (more) load_global(k0 + BK);   // overlap global fetch with WMMA

    v16bf af[2], b1f[2], b3f[2];
#pragma unroll
    for (int tm = 0; tm < 2; ++tm) af[tm] = load_frag(&As[p][wm * 32 + tm * 16][0]);
#pragma unroll
    for (int tn = 0; tn < 2; ++tn) {
      b1f[tn] = load_frag(&Bs1[p][wn * 32 + tn * 16][0]);
      b3f[tn] = load_frag(&Bs3[p][wn * 32 + tn * 16][0]);
    }
#pragma unroll
    for (int tm = 0; tm < 2; ++tm)
#pragma unroll
      for (int tn = 0; tn < 2; ++tn) {
        acc1[tm][tn] = __builtin_amdgcn_wmma_f32_16x16x32_bf16(
            false, af[tm], false, b1f[tn], (short)0, acc1[tm][tn], false, false);
        acc3[tm][tn] = __builtin_amdgcn_wmma_f32_16x16x32_bf16(
            false, af[tm], false, b3f[tn], (short)0, acc3[tm][tn], false, false);
      }

    if (more) store_lds(p ^ 1);       // fill the other buffer
    __syncthreads();
    p ^= 1;
  }

  // Epilogue: h = silu(a1) * a3, bf16. C/D layout: VGPR v -> M = v (+8 hi lanes).
  const int mhi = (lane & 16) ? 8 : 0;
#pragma unroll
  for (int tm = 0; tm < 2; ++tm)
#pragma unroll
    for (int tn = 0; tn < 2; ++tn)
#pragma unroll
      for (int v = 0; v < 8; ++v) {
        float a1 = acc1[tm][tn][v];
        float a3 = acc3[tm][tn][v];
        float hv = (a1 / (1.f + __expf(-a1))) * a3;
        int row = m0 + wm * 32 + tm * 16 + mhi + v;
        int col = n0 + wn * 32 + tn * 16 + (lane & 15);
        H[(size_t)row * kHID + col] = (__bf16)hv;
      }
}

// ---------------------------------------------------------------------------
// FFN second stage: Out = H @ W2. Block tile 128x128, wave tile 32x64
// (2x4 WMMA accumulators -> 8 wmma per K-step per wave), double-buffered.
// Shared mode stores; routed mode scatter-adds via global_atomic_add_f32.
// ---------------------------------------------------------------------------
__global__ __launch_bounds__(256)
void gemm2_kernel(const __bf16* __restrict__ H,
                  const float* __restrict__ W2base,
                  const int* __restrict__ row_tok,   // null => shared (store)
                  const int* __restrict__ offs,      // null => single expert
                  float* __restrict__ Out) {
  __shared__ __align__(16) __bf16 As[2][BM][LDT];
  __shared__ __align__(16) __bf16 Bs[2][GN][LDT];

  const int m0 = blockIdx.x * BM;
  const int n0 = blockIdx.y * GN;

  const float* W2 = W2base;
  if (offs) {
    if (m0 >= offs[kE]) return;
    int e = 0;
    while (e + 1 < kE && m0 >= offs[e + 1]) ++e;
    W2 += (size_t)e * kHID * kDIM;
  }

  const int tid  = threadIdx.x;
  const int lane = tid & 31;
  const int wid  = tid >> 5;
  const int wm   = wid >> 1;    // 0..3 -> 32-row slice
  const int wn   = wid & 1;     // 0..1 -> 64-col slice

  v8f acc[2][4];
#pragma unroll
  for (int i = 0; i < 2; ++i)
#pragma unroll
    for (int j = 0; j < 4; ++j) { v8f z = {}; acc[i][j] = z; }

  const int ar = tid >> 1;
  const int ac = (tid & 1) * 16;

  v8bf  aReg[2];
  float4 bReg[4];

  auto load_global = [&](int k0) {
    const v8bf* src = (const v8bf*)(H + (size_t)(m0 + ar) * kHID + k0 + ac);
    aReg[0] = src[0];
    aReg[1] = src[1];
#pragma unroll
    for (int f = 0; f < 4; ++f) {
      int fi = tid * 4 + f;            // 0..1023
      int bk = fi >> 5;                // 0..31
      int bn = (fi & 31) * 4;          // 0..124
      const float* s2 = W2 + (size_t)(k0 + bk) * kDIM + n0 + bn;
      bReg[f] = *(const float4*)s2;
      __builtin_prefetch(s2 + (size_t)BK * kDIM, 0, 1);
    }
  };
  auto store_lds = [&](int p) {
    *(v8bf*)&As[p][ar][ac]     = aReg[0];
    *(v8bf*)&As[p][ar][ac + 8] = aReg[1];
#pragma unroll
    for (int f = 0; f < 4; ++f) {
      int fi = tid * 4 + f;
      int bk = fi >> 5;
      int bn = (fi & 31) * 4;
      Bs[p][bn + 0][bk] = (__bf16)bReg[f].x; Bs[p][bn + 1][bk] = (__bf16)bReg[f].y;
      Bs[p][bn + 2][bk] = (__bf16)bReg[f].z; Bs[p][bn + 3][bk] = (__bf16)bReg[f].w;
    }
  };

  load_global(0);
  store_lds(0);
  __syncthreads();

  int p = 0;
  for (int k0 = 0; k0 < kHID; k0 += BK) {
    const bool more = (k0 + BK) < kHID;
    if (more) load_global(k0 + BK);

    v16bf af[2], bf[4];
#pragma unroll
    for (int tm = 0; tm < 2; ++tm) af[tm] = load_frag(&As[p][wm * 32 + tm * 16][0]);
#pragma unroll
    for (int tn = 0; tn < 4; ++tn) bf[tn] = load_frag(&Bs[p][wn * 64 + tn * 16][0]);
#pragma unroll
    for (int tm = 0; tm < 2; ++tm)
#pragma unroll
      for (int tn = 0; tn < 4; ++tn)
        acc[tm][tn] = __builtin_amdgcn_wmma_f32_16x16x32_bf16(
            false, af[tm], false, bf[tn], (short)0, acc[tm][tn], false, false);

    if (more) store_lds(p ^ 1);
    __syncthreads();
    p ^= 1;
  }

  const int mhi = (lane & 16) ? 8 : 0;
#pragma unroll
  for (int tm = 0; tm < 2; ++tm)
#pragma unroll
    for (int tn = 0; tn < 4; ++tn)
#pragma unroll
      for (int v = 0; v < 8; ++v) {
        float val = acc[tm][tn][v];
        int row = m0 + wm * 32 + tm * 16 + mhi + v;
        int col = n0 + wn * 64 + tn * 16 + (lane & 15);
        if (row_tok) {
          int tok = row_tok[row];
          if (tok >= 0) atomicAdd(&Out[(size_t)tok * kDIM + col], val);
        } else {
          Out[(size_t)row * kDIM + col] = val;
        }
      }
}

// ---------------------------------------------------------------------------
// Launch
// ---------------------------------------------------------------------------
extern "C" void kernel_launch(void* const* d_in, const int* in_sizes, int n_in,
                              void* d_out, int out_size, void* d_ws, size_t ws_size,
                              hipStream_t stream) {
  const float* x      = (const float*)d_in[0];
  const float* gate_w = (const float*)d_in[1];
  const float* w1     = (const float*)d_in[2];
  const float* w2     = (const float*)d_in[3];
  const float* w3     = (const float*)d_in[4];
  const float* sw1    = (const float*)d_in[5];
  const float* sw2    = (const float*)d_in[6];
  const float* sw3    = (const float*)d_in[7];
  const float* bias   = (const float*)d_in[8];
  float* out = (float*)d_out;

  char* ws = (char*)d_ws;
  auto take = [&](size_t bytes) -> char* {
    char* p = ws;
    ws += (bytes + 255) & ~(size_t)255;
    return p;
  };
  int*    counts  = (int*)   take(kE * sizeof(int));
  int*    cursor  = (int*)   take(kE * sizeof(int));
  int*    offs    = (int*)   take((kE + 1) * sizeof(int));
  int*    sel_e   = (int*)   take((size_t)kRR * sizeof(int));
  float*  sel_s   = (float*) take((size_t)kRR * sizeof(float));
  int*    row_tok = (int*)   take((size_t)kMAXPAD * sizeof(int));
  float*  row_scl = (float*) take((size_t)kMAXPAD * sizeof(float));
  __bf16* Hbuf    = (__bf16*)take((size_t)kMAXPAD * kHID * sizeof(__bf16));

  // Routing metadata.
  init_kernel  <<<(kMAXPAD + 255) / 256, 256, 0, stream>>>(row_tok, counts);
  router_kernel<<<(kT + 255) / 256,      256, 0, stream>>>(x, gate_w, bias, sel_e, sel_s, counts);
  scan_kernel  <<<1, 32, 0, stream>>>(counts, offs, cursor);
  assign_kernel<<<(kRR + 255) / 256,     256, 0, stream>>>(sel_e, sel_s, offs, cursor, row_tok, row_scl);

  // Shared expert: writes every element of out (no pre-zero needed).
  ffn13_kernel<<<dim3(kT / BM, kHID / BN), 256, 0, stream>>>(
      x, sw1, sw3, nullptr, nullptr, nullptr, Hbuf);
  gemm2_kernel<<<dim3(kT / BM, kDIM / GN), 256, 0, stream>>>(
      Hbuf, sw2, nullptr, nullptr, out);

  // Routed experts: grouped ragged GEMMs, scatter-add into out.
  ffn13_kernel<<<dim3(kMAXPAD / BM, kHID / BN), 256, 0, stream>>>(
      x, w1, w3, row_tok, row_scl, offs, Hbuf);
  gemm2_kernel<<<dim3(kMAXPAD / BM, kDIM / GN), 256, 0, stream>>>(
      Hbuf, w2, row_tok, offs, out);
}
